// Mamba2Backbone_38328288149700
// MI455X (gfx1250) — matrix operand order repaired and checked by reading, not compiled, not run on hardware
//
#include <hip/hip_runtime.h>
#include <hip/hip_bf16.h>

#define B_ROWS 16384
#define S_DIM  1024
#define A_DIM  512
#define IN_DIM 1536
#define HID    2048
#define DET    4096
#define EPSN   1e-4f

// WG tile: 128 (M) x 64 (N), 256 threads = 8 waves in 4x2, wave tile 32x32.
#define TILE_M 128
#define TILE_N 64
#define LDS_ROW 80                       // 64B of k-data + 16B pad (bank spread)
#define A_TILE_BYTES (TILE_M * LDS_ROW)  // 10240
#define B_TILE_BYTES (TILE_N * LDS_ROW)  // 5120

typedef __attribute__((ext_vector_type(16))) __bf16 v16bf;
typedef __attribute__((ext_vector_type(8)))  __bf16 v8bf;
typedef __attribute__((ext_vector_type(8)))  float  v8f;
typedef __attribute__((ext_vector_type(4)))  float  v4f;
typedef __attribute__((ext_vector_type(4)))  int    v4i;

#define WMMA_BF16(a, b, c) \
  __builtin_amdgcn_wmma_f32_16x16x32_bf16(false, (a), false, (b), (short)0, (c), false, false)

#if __has_builtin(__builtin_amdgcn_global_load_async_to_lds_b128)
#define HAVE_ASYNC_LDS 1
#else
#define HAVE_ASYNC_LDS 0
#endif

// 16-byte global -> LDS copy (async if the toolchain has the gfx1250 builtin)
static __device__ __forceinline__ void copy16_g2l(const void* g, void* l) {
#if HAVE_ASYNC_LDS
  __builtin_amdgcn_global_load_async_to_lds_b128(
      (__attribute__((address_space(1))) v4i*)(g),
      (__attribute__((address_space(3))) v4i*)(l), 0, 0);
#else
  *(v4f*)l = *(const v4f*)g;
#endif
}

static __device__ __forceinline__ void wait_stage_barrier() {
#if HAVE_ASYNC_LDS
#if __has_builtin(__builtin_amdgcn_s_wait_asynccnt)
  __builtin_amdgcn_s_wait_asynccnt(0);
#else
  asm volatile("s_wait_asynccnt 0" ::: "memory");
#endif
#endif
  __syncthreads();
}

// ---- WMMA fragment loads from LDS tiles (ISA 16-bit A 16x32 / B 32x16 layouts) --
// A tile in LDS: 128 rows (m) x 32 k, row stride LDS_ROW bytes.
static __device__ __forceinline__ v16bf lds_a_frag(const char* base, int mrow, int lane) {
  const int hl = lane >> 4, r = lane & 15;
  const char* p = base + (mrow + r) * LDS_ROW + hl * 16;
  union { v16bf v; v8bf h[2]; } u;
  u.h[0] = *(const v8bf*)(p);        // k = 8*hl .. 8*hl+7
  u.h[1] = *(const v8bf*)(p + 32);   // k = 16+8*hl .. 16+8*hl+7
  return u.v;
}
// B tile in LDS: 64 rows (n) x 32 k, row stride LDS_ROW bytes.
static __device__ __forceinline__ v16bf lds_b_frag(const char* base, int nrow, int lane) {
  const int hl = lane >> 4, r = lane & 15;
  const char* p = base + (nrow + r) * LDS_ROW + hl * 16;
  union { v16bf v; v8bf h[2]; } u;
  u.h[0] = *(const v8bf*)(p);        // k = 8*hl .. 8*hl+7
  u.h[1] = *(const v8bf*)(p + 32);   // k = 16+8*hl .. 16+8*hl+7
  return u.v;
}

static __device__ __forceinline__ float softplusf(float x) {
  return (x > 20.f) ? x : log1pf(__expf(x));
}
static __device__ __forceinline__ float sigmoidf(float x) {
  return 1.f / (1.f + __expf(-x));
}

// ---- LDS-staged, double-buffered multi-matrix GEMM core -----------------------
// C_t[m,n] = sum_k A[m,k] * W_t[n,k], accumulators per wave: 32x32 per matrix.
template <int NMAT>
static __device__ __forceinline__ void gemm_lds_core(
    const __bf16* __restrict__ A,
    const __bf16* __restrict__ W0, const __bf16* __restrict__ W1,
    const __bf16* __restrict__ W2, const __bf16* __restrict__ W3,
    int K, char* smem, v8f (&acc)[NMAT][2][2]) {
  const int tid  = threadIdx.x;
  const int lane = tid & 31;
  const int wave = tid >> 5;
  const int wm = wave & 3, wn = wave >> 2;
  const size_t mr_wg = (size_t)blockIdx.x * TILE_M;
  const size_t nb_wg = (size_t)blockIdx.y * TILE_N;
  const __bf16* Wm[4] = {W0, W1, W2, W3};
  const int BUF = A_TILE_BYTES + NMAT * B_TILE_BYTES;

#pragma unroll
  for (int t = 0; t < NMAT; ++t)
#pragma unroll
    for (int i = 0; i < 2; ++i)
#pragma unroll
      for (int j = 0; j < 2; ++j) acc[t][i][j] = {};

  // cooperative stage of one 32-wide K slice into LDS buffer `buf`
  auto stage = [&](int buf, int k0) {
    char* base = smem + buf * BUF;
    // A: 128 rows x 64B = 512 x 16B chunks, 2 per thread
#pragma unroll
    for (int c = 0; c < 2; ++c) {
      const int cc = tid + c * 256;
      const int row = cc >> 2, q = cc & 3;
      const __bf16* g = A + (mr_wg + row) * (size_t)K + k0 + q * 8;
      char* l = base + row * LDS_ROW + q * 16;
      copy16_g2l(g, l);
    }
    // B: NMAT mats x 64 rows x 64B = NMAT x 256 x 16B chunks, NMAT per thread
#pragma unroll
    for (int t = 0; t < NMAT; ++t) {
      const int row = tid >> 2, q = tid & 3;
      const __bf16* g = Wm[t] + (nb_wg + row) * (size_t)K + k0 + q * 8;
      char* l = base + A_TILE_BYTES + t * B_TILE_BYTES + row * LDS_ROW + q * 16;
      copy16_g2l(g, l);
    }
  };

  auto compute = [&](int buf) {
    char* base = smem + buf * BUF;
    v16bf a0 = lds_a_frag(base, wm * 32,      lane);
    v16bf a1 = lds_a_frag(base, wm * 32 + 16, lane);
#pragma unroll
    for (int t = 0; t < NMAT; ++t) {
      char* bb = base + A_TILE_BYTES + t * B_TILE_BYTES;
#pragma unroll
      for (int in = 0; in < 2; ++in) {
        v16bf b = lds_b_frag(bb, wn * 32 + in * 16, lane);
        acc[t][0][in] = WMMA_BF16(a0, b, acc[t][0][in]);
        acc[t][1][in] = WMMA_BF16(a1, b, acc[t][1][in]);
      }
    }
  };

  const int nk = K / 32;
  stage(0, 0);
  wait_stage_barrier();
  for (int i = 0; i < nk; ++i) {
    const int cur = i & 1;
    if (i + 1 < nk) stage(cur ^ 1, (i + 1) * 32);  // prefetch next slice
    compute(cur);                                   // WMMA on current slice
    wait_stage_barrier();                           // own async done + all waves synced
  }
}

// ---- prep kernels -------------------------------------------------------------
__global__ void k_f32_to_bf16(const float* __restrict__ src, __bf16* __restrict__ dst, int n) {
  int i = blockIdx.x * blockDim.x + threadIdx.x;
  if (i < n) dst[i] = (__bf16)src[i];
}

__global__ void k_prep_u(const float* __restrict__ stoch, const float* __restrict__ action,
                         __bf16* __restrict__ u) {
  int i = blockIdx.x * blockDim.x + threadIdx.x;
  if (i >= B_ROWS * IN_DIM) return;
  int m = i / IN_DIM, c = i % IN_DIM;
  float v;
  if (c < S_DIM) {
    v = stoch[(size_t)m * S_DIM + c];
  } else {
    float a = action[(size_t)m * A_DIM + (c - S_DIM)];
    v = a / fmaxf(fabsf(a), 1.0f);
  }
  u[i] = (__bf16)v;
}

// ---- GEMM1: h_pre = u @ w1.T + b1 (fp32 out) ----------------------------------
__global__ __launch_bounds__(256) void k_gemm1(
    const __bf16* __restrict__ A, const __bf16* __restrict__ W,
    const float* __restrict__ bias, int N, int K, float* __restrict__ outF) {
  __shared__ __align__(16) char smem[2 * (A_TILE_BYTES + 1 * B_TILE_BYTES)];
  v8f acc[1][2][2];
  gemm_lds_core<1>(A, W, W, W, W, K, smem, acc);

  const int lane = threadIdx.x & 31, wave = threadIdx.x >> 5;
  const int wm = wave & 3, wn = wave >> 2;
  const int mr = blockIdx.x * TILE_M + wm * 32;
  const int nb = blockIdx.y * TILE_N + wn * 32;
  const int hl = lane >> 4, nl = lane & 15;
#pragma unroll
  for (int im = 0; im < 2; ++im)
#pragma unroll
    for (int in = 0; in < 2; ++in) {
      const int n = nb + in * 16 + nl;
      const float bs = bias[n];
#pragma unroll
      for (int j = 0; j < 8; ++j) {
        const int m = mr + im * 16 + hl * 8 + j;
        outF[(size_t)m * N + n] = acc[0][im][in][j] + bs;
      }
    }
}

// ---- GEMM2: x = h @ w2.T + b2 (fp32 + bf16 mirror) ----------------------------
__global__ __launch_bounds__(256) void k_gemm2(
    const __bf16* __restrict__ A, const __bf16* __restrict__ W,
    const float* __restrict__ bias, int N, int K,
    float* __restrict__ outF, __bf16* __restrict__ outBF) {
  __shared__ __align__(16) char smem[2 * (A_TILE_BYTES + 1 * B_TILE_BYTES)];
  v8f acc[1][2][2];
  gemm_lds_core<1>(A, W, W, W, W, K, smem, acc);

  const int lane = threadIdx.x & 31, wave = threadIdx.x >> 5;
  const int wm = wave & 3, wn = wave >> 2;
  const int mr = blockIdx.x * TILE_M + wm * 32;
  const int nb = blockIdx.y * TILE_N + wn * 32;
  const int hl = lane >> 4, nl = lane & 15;
#pragma unroll
  for (int im = 0; im < 2; ++im)
#pragma unroll
    for (int in = 0; in < 2; ++in) {
      const int n = nb + in * 16 + nl;
      const float bs = bias[n];
#pragma unroll
      for (int j = 0; j < 8; ++j) {
        const int m = mr + im * 16 + hl * 8 + j;
        const float v = acc[0][im][in][j] + bs;
        const size_t idx = (size_t)m * N + n;
        outF[idx]  = v;
        outBF[idx] = (__bf16)v;
      }
    }
}

// ---- h = silu(rmsnorm(h_pre, tok_nw)) -> bf16, one row per block ---------------
__global__ __launch_bounds__(256) void k_rms_silu(const float* __restrict__ hp,
                                                  const float* __restrict__ nw,
                                                  __bf16* __restrict__ hb) {
  const int row = blockIdx.x;
  const float* r = hp + (size_t)row * HID;
  float v[8];
  float ss = 0.f;
#pragma unroll
  for (int i = 0; i < 8; ++i) { v[i] = r[threadIdx.x + i * 256]; ss += v[i] * v[i]; }
#pragma unroll
  for (int o = 16; o > 0; o >>= 1) ss += __shfl_xor(ss, o, 32);
  __shared__ float sred[8];
  if ((threadIdx.x & 31) == 0) sred[threadIdx.x >> 5] = ss;
  __syncthreads();
  float tot = 0.f;
#pragma unroll
  for (int i = 0; i < 8; ++i) tot += sred[i];
  const float inv = rsqrtf(tot * (1.0f / HID) + EPSN);
#pragma unroll
  for (int i = 0; i < 8; ++i) {
    const int c = threadIdx.x + i * 256;
    const float s = v[i] * inv * nw[c];
    hb[(size_t)row * HID + c] = (__bf16)(s * sigmoidf(s));
  }
}

// ---- fused quad GEMM (dt,b,c,z share the A fragments) + SSM epilogue -----------
__global__ __launch_bounds__(256) void k_quad_gemm(
    const __bf16* __restrict__ Xb,
    const __bf16* __restrict__ Wdt, const __bf16* __restrict__ Wb,
    const __bf16* __restrict__ Wc,  const __bf16* __restrict__ Wz,
    const float* __restrict__ dt_b, const float* __restrict__ b_b,
    const float* __restrict__ c_b,  const float* __restrict__ z_b,
    const float* __restrict__ a_base, const float* __restrict__ skip,
    const float* __restrict__ Xf, const float* __restrict__ deter,
    float* __restrict__ out) {
  __shared__ __align__(16) char smem[2 * (A_TILE_BYTES + 4 * B_TILE_BYTES)];
  v8f acc[4][2][2];
  gemm_lds_core<4>(Xb, Wdt, Wb, Wc, Wz, DET, smem, acc);

  const int lane = threadIdx.x & 31, wave = threadIdx.x >> 5;
  const int wm = wave & 3, wn = wave >> 2;
  const int mr = blockIdx.x * TILE_M + wm * 32;
  const int nb = blockIdx.y * TILE_N + wn * 32;
  const int hl = lane >> 4, nl = lane & 15;
#pragma unroll
  for (int im = 0; im < 2; ++im) {
#pragma unroll
    for (int in = 0; in < 2; ++in) {
      const int n = nb + in * 16 + nl;
      const float dtb = dt_b[n], bb = b_b[n], cb = c_b[n], zb = z_b[n];
      const float an = -softplusf(a_base[n]);
      const float sk = skip[n];
#pragma unroll
      for (int j = 0; j < 8; ++j) {
        const int m = mr + im * 16 + hl * 8 + j;
        const size_t idx = (size_t)m * DET + n;
        const float xv = Xf[idx];
        const float dv = deter[idx];
        const float dt = softplusf(acc[0][im][in][j] + dtb) + 1e-4f;
        const float bv = tanhf(acc[1][im][in][j] + bb);
        const float cv = acc[2][im][in][j] + cb;
        const float zv = sigmoidf(acc[3][im][in][j] + zb);
        const float state = __expf(an * dt) * dv + dt * bv * xv;
        const float y = cv * state + sk * xv;
        out[idx] = zv * y + (1.f - zv) * dv;
      }
    }
  }
}

// ---- final in-place RMSNorm over DET per row -----------------------------------
__global__ __launch_bounds__(256) void k_final_norm(float* __restrict__ out,
                                                    const float* __restrict__ nw) {
  const int row = blockIdx.x;
  float* r = out + (size_t)row * DET;
  float v[16];
  float ss = 0.f;
#pragma unroll
  for (int i = 0; i < 16; ++i) { v[i] = r[threadIdx.x + i * 256]; ss += v[i] * v[i]; }
#pragma unroll
  for (int o = 16; o > 0; o >>= 1) ss += __shfl_xor(ss, o, 32);
  __shared__ float sred[8];
  if ((threadIdx.x & 31) == 0) sred[threadIdx.x >> 5] = ss;
  __syncthreads();
  float tot = 0.f;
#pragma unroll
  for (int i = 0; i < 8; ++i) tot += sred[i];
  const float inv = rsqrtf(tot * (1.0f / DET) + EPSN);
#pragma unroll
  for (int i = 0; i < 16; ++i) {
    const int c = threadIdx.x + i * 256;
    r[c] = v[i] * inv * nw[c];
  }
}

extern "C" void kernel_launch(void* const* d_in, const int* in_sizes, int n_in,
                              void* d_out, int out_size, void* d_ws, size_t ws_size,
                              hipStream_t stream) {
  const float* stoch  = (const float*)d_in[0];
  const float* deter  = (const float*)d_in[1];
  const float* action = (const float*)d_in[2];
  const float* tok_w1 = (const float*)d_in[3];
  const float* tok_b1 = (const float*)d_in[4];
  const float* tok_nw = (const float*)d_in[5];
  const float* tok_w2 = (const float*)d_in[6];
  const float* tok_b2 = (const float*)d_in[7];
  const float* a_base = (const float*)d_in[8];
  const float* dt_w   = (const float*)d_in[9];
  const float* dt_b   = (const float*)d_in[10];
  const float* b_w    = (const float*)d_in[11];
  const float* b_b    = (const float*)d_in[12];
  const float* c_w    = (const float*)d_in[13];
  const float* c_b    = (const float*)d_in[14];
  const float* z_w    = (const float*)d_in[15];
  const float* z_b    = (const float*)d_in[16];
  const float* skip   = (const float*)d_in[17];
  const float* norm_w = (const float*)d_in[18];

  char* ws = (char*)d_ws;
  size_t off = 0;
  auto alloc = [&](size_t bytes) -> char* {
    char* p = ws + off;
    off += (bytes + 255) & ~(size_t)255;
    return p;
  };
  __bf16* u_bf   = (__bf16*)alloc((size_t)B_ROWS * IN_DIM * 2);
  __bf16* w1_bf  = (__bf16*)alloc((size_t)HID * IN_DIM * 2);
  __bf16* w2_bf  = (__bf16*)alloc((size_t)DET * HID * 2);
  __bf16* wdt_bf = (__bf16*)alloc((size_t)DET * DET * 2);
  __bf16* wb_bf  = (__bf16*)alloc((size_t)DET * DET * 2);
  __bf16* wc_bf  = (__bf16*)alloc((size_t)DET * DET * 2);
  __bf16* wz_bf  = (__bf16*)alloc((size_t)DET * DET * 2);
  float*  h_pre  = (float*)alloc((size_t)B_ROWS * HID * 4);
  __bf16* h_bf   = (__bf16*)alloc((size_t)B_ROWS * HID * 2);
  float*  x_f    = (float*)alloc((size_t)B_ROWS * DET * 4);
  __bf16* x_bf   = (__bf16*)alloc((size_t)B_ROWS * DET * 2);

  auto cvt = [&](const float* s, __bf16* d, size_t n) {
    k_f32_to_bf16<<<(unsigned)((n + 255) / 256), 256, 0, stream>>>(s, d, (int)n);
  };
  cvt(tok_w1, w1_bf, (size_t)HID * IN_DIM);
  cvt(tok_w2, w2_bf, (size_t)DET * HID);
  cvt(dt_w, wdt_bf, (size_t)DET * DET);
  cvt(b_w,  wb_bf,  (size_t)DET * DET);
  cvt(c_w,  wc_bf,  (size_t)DET * DET);
  cvt(z_w,  wz_bf,  (size_t)DET * DET);

  k_prep_u<<<(B_ROWS * IN_DIM + 255) / 256, 256, 0, stream>>>(stoch, action, u_bf);

  dim3 g1(B_ROWS / TILE_M, HID / TILE_N);
  k_gemm1<<<g1, 256, 0, stream>>>(u_bf, w1_bf, tok_b1, HID, IN_DIM, h_pre);

  k_rms_silu<<<B_ROWS, 256, 0, stream>>>(h_pre, tok_nw, h_bf);

  dim3 g2(B_ROWS / TILE_M, DET / TILE_N);
  k_gemm2<<<g2, 256, 0, stream>>>(h_bf, w2_bf, tok_b2, DET, HID, x_f, x_bf);

  dim3 g3(B_ROWS / TILE_M, DET / TILE_N);
  k_quad_gemm<<<g3, 256, 0, stream>>>(x_bf, wdt_bf, wb_bf, wc_bf, wz_bf,
                                      dt_b, b_b, c_b, z_b, a_base, skip,
                                      x_f, deter, (float*)d_out);

  k_final_norm<<<B_ROWS, 256, 0, stream>>>((float*)d_out, norm_w);
}
